// DRSATemporalBlock_55138790146508
// MI455X (gfx1250) — compile-verified, matrix operand order checked by reading, not run on hardware
//
#include <hip/hip_runtime.h>
#include <hip/hip_bf16.h>

// ---------------------------------------------------------------------------
// Types for CDNA5 WMMA (wave32)
// ---------------------------------------------------------------------------
typedef __attribute__((ext_vector_type(16))) __bf16 v16bf;
typedef __attribute__((ext_vector_type(8)))  float  v8f;

union Frag {
    unsigned int   u[8];
    unsigned short h[16];
    v16bf          v;
};

// 16-bit A/B fragment K-striping (ISA 7.12.2, 16-bit A 16x32):
//  lane<16 : pairs v<4 -> K=2v,2v+1 ; v>=4 -> K=2v+8,2v+9
//  lane>=16: same +8
__device__ __forceinline__ int kpair(int v, int half) {
    return ((v < 4) ? 2 * v : 2 * v + 8) + 8 * half;
}
__device__ __forceinline__ int kelem(int j, int half) {
    return kpair(j >> 1, half) + (j & 1);
}

__device__ __forceinline__ unsigned short f2bf(float f) {
    unsigned int u = __float_as_uint(f);
    unsigned int r = u + 0x7FFFu + ((u >> 16) & 1u);   // round-to-nearest-even
    return (unsigned short)(r >> 16);
}
__device__ __forceinline__ float bf2f(unsigned short h) {
    return __uint_as_float(((unsigned int)h) << 16);
}

// ---------------------------------------------------------------------------
// Problem constants
// ---------------------------------------------------------------------------
#define BB   2
#define HH   64
#define WW   24
#define CC   256
#define PP   8
#define KV   4
#define NHH  8
#define DK   32
#define NTOK 3072       // B*H*W
#define LROW 192        // hp*W tokens per block

// ---------------------------------------------------------------------------
// fp32 -> bf16 conversion (grid-stride)
// ---------------------------------------------------------------------------
__global__ void cvt_bf16_kernel(const float* __restrict__ src,
                                unsigned short* __restrict__ dst, int n) {
    for (int i = blockIdx.x * blockDim.x + threadIdx.x; i < n;
         i += gridDim.x * blockDim.x)
        dst[i] = f2bf(src[i]);
}

// out = bf16(a + b), elementwise
__global__ void add_bf16_kernel(const unsigned short* __restrict__ a,
                                const unsigned short* __restrict__ b,
                                unsigned short* __restrict__ out, int n) {
    for (int i = blockIdx.x * blockDim.x + threadIdx.x; i < n;
         i += gridDim.x * blockDim.x)
        out[i] = f2bf(bf2f(a[i]) + bf2f(b[i]));
}

// ---------------------------------------------------------------------------
// Pack a [256,256] K-major f32 weight into B-fragment order:
//   P[(((nt*8)+kb)*32 + lane)*16 + j] = bf16( W[(kb*32+kelem(j,half))*256 + nt*16+lm] )
// 65536 elements, grid 256x256.
// ---------------------------------------------------------------------------
__global__ void pack_wB_kernel(const float* __restrict__ W,
                               unsigned short* __restrict__ P) {
    const int idx  = blockIdx.x * 256 + threadIdx.x;
    const int j    = idx & 15;
    const int lane = (idx >> 4) & 31;
    const int kb   = (idx >> 9) & 7;
    const int nt   = idx >> 12;
    const int half = lane >> 4, lm = lane & 15;
    const int k = kb * 32 + kelem(j, half);
    const int n = nt * 16 + lm;
    P[idx] = f2bf(W[(size_t)k * CC + n]);
}

// ---------------------------------------------------------------------------
// Pack conv HWIO weights [5,5,256,256] into B-fragment order:
//   P[((((tap*8)+kb)*16 + nt)*32 + lane)*16 + j]
// 1638400 elements, grid 6400x256.
// ---------------------------------------------------------------------------
__global__ void pack_cwB_kernel(const float* __restrict__ W,
                                unsigned short* __restrict__ P) {
    const int idx  = blockIdx.x * 256 + threadIdx.x;
    const int j    = idx & 15;
    const int lane = (idx >> 4) & 31;
    const int nt   = (idx >> 9) & 15;
    const int kbt  = idx >> 13;            // tap*8 + kb, 0..199
    const int kb   = kbt & 7;
    const int tap  = kbt >> 3;
    const int half = lane >> 4, lm = lane & 15;
    const int cin  = kb * 32 + kelem(j, half);
    const int cout = nt * 16 + lm;
    P[idx] = f2bf(W[((size_t)tap * CC + cin) * CC + cout]);
}

// ---------------------------------------------------------------------------
// Transpose va (bf16) to d-major: vat[(blk16*256 + c)*192 + s] = va[(blk16*192+s)*256 + c]
// ---------------------------------------------------------------------------
__global__ void transpose_va_kernel(const unsigned short* __restrict__ va,
                                    unsigned short* __restrict__ vat) {
    const int idx = blockIdx.x * 256 + threadIdx.x;        // < 786432
    const int s = idx % LROW;
    const int rest = idx / LROW;
    const int c = rest & 255;
    const int blk16 = rest >> 8;                           // 0..15
    vat[idx] = va[((size_t)blk16 * LROW + s) * CC + c];
}

// ---------------------------------------------------------------------------
// GEMM: Y[M,256] = X[M,256] @ W[256,256] + bias   (bf16 in, f32 acc)
// B operand pre-packed in fragment order. One 16x16 tile/wave, grid (M/64,16).
// ---------------------------------------------------------------------------
__global__ void __launch_bounds__(128)
gemm256_kernel(const unsigned short* __restrict__ X,
               const unsigned short* __restrict__ Wp,
               const float* __restrict__ bias,
               unsigned short* __restrict__ Ybf,
               float* __restrict__ Yf) {
    const int wave = threadIdx.x >> 5, lane = threadIdx.x & 31;
    const int half = lane >> 4, lm = lane & 15;
    const int mt = blockIdx.x * 4 + wave;
    const int nt = blockIdx.y;
    const int row = mt * 16 + lm;      // A row (same for both halves)
    const int col = nt * 16 + lm;      // B/D column

    v8f acc = {};
#pragma unroll
    for (int kb = 0; kb < 8; ++kb) {
        Frag a;
#pragma unroll
        for (int v = 0; v < 8; ++v)
            a.u[v] = *(const unsigned int*)(X + (size_t)row * CC + kb * 32 +
                                            kpair(v, half));
        const v16bf bv =
            *(const v16bf*)(Wp + (((size_t)nt * 8 + kb) * 32 + lane) * 16);
        acc = __builtin_amdgcn_wmma_f32_16x16x32_bf16(false, a.v, false, bv,
                                                      (short)0, acc, false, false);
    }
    const float bs = bias[col];
#pragma unroll
    for (int r = 0; r < 8; ++r) {
        const int orow = mt * 16 + r + 8 * half;
        const size_t idx = (size_t)orow * CC + col;
        const float val = acc[r] + bs;
        if (Yf)  Yf[idx] = val;
        if (Ybf) Ybf[idx] = f2bf(val);
    }
}

// ---------------------------------------------------------------------------
// Routing: block descriptors (means over 192 tokens) then top-4 selection
// ---------------------------------------------------------------------------
__global__ void block_means_kernel(const unsigned short* __restrict__ qbf,
                                   const unsigned short* __restrict__ kbf,
                                   float* __restrict__ qloc,
                                   float* __restrict__ kloc) {
    const int bp = blockIdx.x;         // 0..15
    const int c = threadIdx.x;         // 0..255
    float sq = 0.f, sk = 0.f;
    const size_t base = (size_t)bp * LROW * CC + c;
    for (int r = 0; r < LROW; ++r) {
        sq += bf2f(qbf[base + (size_t)r * CC]);
        sk += bf2f(kbf[base + (size_t)r * CC]);
    }
    qloc[bp * CC + c] = sq * (1.f / LROW);
    kloc[bp * CC + c] = sk * (1.f / LROW);
}

__global__ void routing_topk_kernel(const float* __restrict__ qloc,
                                    const float* __restrict__ kloc,
                                    int* __restrict__ R) {
    const int lane = threadIdx.x;
    if (lane >= BB * PP) return;
    const int b = lane >> 3;
    float s[PP];
    for (int q = 0; q < PP; ++q) {
        float acc = 0.f;
        for (int c = 0; c < CC; ++c)
            acc += qloc[lane * CC + c] * kloc[(b * PP + q) * CC + c];
        s[q] = acc;
    }
    bool used[PP] = {};
    for (int j = 0; j < KV; ++j) {
        int best = 0; float bv = -1e30f;
        for (int q = 0; q < PP; ++q)
            if (!used[q] && s[q] > bv) { bv = s[q]; best = q; }
        used[best] = true;
        R[lane * KV + j] = best;
    }
}

// ---------------------------------------------------------------------------
// Flash-style attention, one wave per (b,p,h,qtile).
// S^T = K_tile(A) x Q^T(B) : lane owns one query column (online softmax).
// P stays in registers: D-layout of S^T and B-layout of P coincide per lane
//   (lane keys {sub*16+r+8*half} == {kelem(j,half)}), remap pb.h[j]=p[j>=8][j&7].
// O^T = V^T(A, from d-major vat) x P(B).
// ---------------------------------------------------------------------------
__global__ void __launch_bounds__(128)
attn_kernel(const unsigned short* __restrict__ qa,
            const unsigned short* __restrict__ ka,
            const unsigned short* __restrict__ vat,
            const int* __restrict__ R,
            unsigned short* __restrict__ ctx) {
    const int wave = threadIdx.x >> 5, lane = threadIdx.x & 31;
    const int half = lane >> 4, lm = lane & 15;
    const int qt = blockIdx.x * 4 + wave;   // 0..11
    const int h  = blockIdx.y;              // 0..7
    const int bp = blockIdx.z;              // 0..15
    const int b  = bp >> 3;
    const float scale = 0.17677669529663687f;   // 1/sqrt(32)

    int Rv[KV];
#pragma unroll
    for (int j = 0; j < KV; ++j) Rv[j] = R[bp * KV + j];

    // Q^T fragment (B operand): column = query lm, element d = kelem(j,half)
    Frag qf;
    const size_t qrow = ((size_t)bp * LROW + qt * 16 + lm) * CC + h * DK;
#pragma unroll
    for (int v = 0; v < 8; ++v)
        qf.u[v] = *(const unsigned int*)(qa + qrow + kpair(v, half));

    // ---- pass 1: per-query online max / sum ----
    float m = -1e30f, l = 0.f;
#pragma unroll 1
    for (int kt = 0; kt < 48; ++kt) {
        const int blk = Rv[kt / 12];
        const int s0 = (kt % 12) * 16;
        const size_t krow = ((size_t)(b * PP + blk) * LROW + s0 + lm) * CC + h * DK;
        Frag af;
#pragma unroll
        for (int v = 0; v < 8; ++v)
            af.u[v] = *(const unsigned int*)(ka + krow + kpair(v, half));
        v8f s = {};
        s = __builtin_amdgcn_wmma_f32_16x16x32_bf16(false, af.v, false, qf.v,
                                                    (short)0, s, false, false);
        float sv[8], mn = m;
#pragma unroll
        for (int r = 0; r < 8; ++r) { sv[r] = s[r] * scale; mn = fmaxf(mn, sv[r]); }
        l *= __expf(m - mn);
        m = mn;
#pragma unroll
        for (int r = 0; r < 8; ++r) l += __expf(sv[r] - m);
    }
    {   // combine the two key-halves (lane L <-> L^16)
        const float mo = __shfl_xor(m, 16, 32);
        const float lo = __shfl_xor(l, 16, 32);
        const float mtot = fmaxf(m, mo);
        l = l * __expf(m - mtot) + lo * __expf(mo - mtot);
        m = mtot;
    }
    const float inv = 1.f / l;

    // ---- pass 2: recompute scores, P in registers, accumulate O^T ----
    v8f o0 = {}, o1 = {};
#pragma unroll 1
    for (int ch = 0; ch < 24; ++ch) {           // 32 keys / chunk
        const int kqi = ch / 6;
        const int blk = Rv[kqi];
        const int sch = ch * 32 - kqi * LROW;   // key offset within block
        float pv[2][8];
#pragma unroll
        for (int sub = 0; sub < 2; ++sub) {
            const int kt = ch * 2 + sub;
            const int s0 = (kt % 12) * 16;
            const size_t krow =
                ((size_t)(b * PP + blk) * LROW + s0 + lm) * CC + h * DK;
            Frag af;
#pragma unroll
            for (int v = 0; v < 8; ++v)
                af.u[v] = *(const unsigned int*)(ka + krow + kpair(v, half));
            v8f s = {};
            s = __builtin_amdgcn_wmma_f32_16x16x32_bf16(false, af.v, false, qf.v,
                                                        (short)0, s, false, false);
#pragma unroll
            for (int r = 0; r < 8; ++r)
                pv[sub][r] = __expf(s[r] * scale - m) * inv;
        }
        Frag pb;                                 // B operand: P [32 keys x 16 q]
#pragma unroll
        for (int r = 0; r < 8; ++r) {
            pb.h[r]     = f2bf(pv[0][r]);
            pb.h[r + 8] = f2bf(pv[1][r]);
        }
        // A operands: V^T d-tiles from d-major vat (contiguous key pairs)
        const size_t vtb =
            ((size_t)(b * PP + blk) * CC + h * DK) * LROW + sch;
        Frag a0, a1;
#pragma unroll
        for (int v = 0; v < 8; ++v) {
            a0.u[v] = *(const unsigned int*)(vat + vtb + (size_t)lm * LROW +
                                             kpair(v, half));
            a1.u[v] = *(const unsigned int*)(vat + vtb + (size_t)(16 + lm) * LROW +
                                             kpair(v, half));
        }
        o0 = __builtin_amdgcn_wmma_f32_16x16x32_bf16(false, a0.v, false, pb.v,
                                                     (short)0, o0, false, false);
        o1 = __builtin_amdgcn_wmma_f32_16x16x32_bf16(false, a1.v, false, pb.v,
                                                     (short)0, o1, false, false);
    }

    // store ctx: lane column = query lm ; D row = d = r + 8*half (+16 for o1)
    const size_t crow = ((size_t)bp * LROW + qt * 16 + lm) * CC + h * DK;
#pragma unroll
    for (int r = 0; r < 8; ++r) {
        ctx[crow + r + 8 * half]      = f2bf(o0[r]);
        ctx[crow + 16 + r + 8 * half] = f2bf(o1[r]);
    }
}

// ---------------------------------------------------------------------------
// 5x5 SAME conv as implicit GEMM with packed weights.
// Grid = (48, 16), 4 waves/block, 200 WMMAs per tile.
// ---------------------------------------------------------------------------
__global__ void __launch_bounds__(128)
conv_gemm_kernel(const unsigned short* __restrict__ kvb,
                 const unsigned short* __restrict__ cwp,
                 const float* __restrict__ bias,
                 float* __restrict__ g) {
    const int wave = threadIdx.x >> 5, lane = threadIdx.x & 31;
    const int half = lane >> 4, lm = lane & 15;
    const int mt = blockIdx.x * 4 + wave;
    const int nt = blockIdx.y;
    const int t = mt * 16 + lm;                 // token for this A row
    const int bb = t / (HH * WW);
    const int rem = t % (HH * WW);
    const int y = rem / WW, x = rem % WW;

    v8f acc = {};
#pragma unroll 1
    for (int tap = 0; tap < 25; ++tap) {
        const int dy = tap / 5 - 2, dx = tap % 5 - 2;
        const int yy = y + dy, xx = x + dx;
        const bool valid = (yy >= 0) & (yy < HH) & (xx >= 0) & (xx < WW);
        const size_t base = ((size_t)(bb * HH + yy) * WW + xx) * CC;
        // prefetch next tap's packed weights (global_prefetch_b8)
        const int ntap = (tap < 24) ? tap + 1 : tap;
        __builtin_prefetch(
            (const void*)(cwp + ((((size_t)ntap * 8) * 16 + nt) * 32 + lane) * 16),
            0, 0);
#pragma unroll
        for (int kb = 0; kb < 8; ++kb) {
            Frag a;
#pragma unroll
            for (int v = 0; v < 8; ++v) {
                unsigned int d =
                    valid ? *(const unsigned int*)(kvb + base + kb * 32 +
                                                   kpair(v, half))
                          : 0u;
                a.u[v] = d;
            }
            const v16bf bv = *(const v16bf*)(
                cwp + ((((size_t)tap * 8 + kb) * 16 + nt) * 32 + lane) * 16);
            acc = __builtin_amdgcn_wmma_f32_16x16x32_bf16(false, a.v, false, bv,
                                                          (short)0, acc, false,
                                                          false);
        }
    }
    const float bs = bias[nt * 16 + lm];
#pragma unroll
    for (int r = 0; r < 8; ++r) {
        const int orow = mt * 16 + r + 8 * half;
        g[(size_t)orow * CC + nt * 16 + lm] = acc[r] + bs;
    }
}

// ---------------------------------------------------------------------------
// out = LayerNorm(a + b) * gamma + beta   (C=256, one block per token row)
// ---------------------------------------------------------------------------
__global__ void ln_add_kernel(const float* __restrict__ A,
                              const float* __restrict__ Bv,
                              const float* __restrict__ gamma,
                              const float* __restrict__ beta,
                              float* __restrict__ out) {
    __shared__ float red[CC];
    const int row = blockIdx.x, c = threadIdx.x;
    const size_t idx = (size_t)row * CC + c;
    const float v = A[idx] + Bv[idx];
    red[c] = v;
    __syncthreads();
    for (int s = 128; s > 0; s >>= 1) {
        if (c < s) red[c] += red[c + s];
        __syncthreads();
    }
    const float mean = red[0] * (1.f / CC);
    __syncthreads();
    const float d = v - mean;
    red[c] = d * d;
    __syncthreads();
    for (int s = 128; s > 0; s >>= 1) {
        if (c < s) red[c] += red[c + s];
        __syncthreads();
    }
    const float var = red[0] * (1.f / CC);
    out[idx] = d * rsqrtf(var + 1e-5f) * gamma[c] + beta[c];
}

// ---------------------------------------------------------------------------
// Launch
// ---------------------------------------------------------------------------
extern "C" void kernel_launch(void* const* d_in, const int* in_sizes, int n_in,
                              void* d_out, int out_size, void* d_ws, size_t ws_size,
                              hipStream_t stream) {
    (void)in_sizes; (void)n_in; (void)out_size; (void)ws_size;
    const float* x      = (const float*)d_in[0];
    const float* wq     = (const float*)d_in[1];
    const float* bq     = (const float*)d_in[2];
    const float* wk     = (const float*)d_in[3];
    const float* bk     = (const float*)d_in[4];
    const float* wv     = (const float*)d_in[5];
    const float* bv     = (const float*)d_in[6];
    const float* wqa    = (const float*)d_in[7];
    const float* bqa    = (const float*)d_in[8];
    const float* wka    = (const float*)d_in[9];
    const float* bka    = (const float*)d_in[10];
    const float* wva    = (const float*)d_in[11];
    const float* bva    = (const float*)d_in[12];
    const float* woa    = (const float*)d_in[13];
    const float* boa    = (const float*)d_in[14];
    const float* convw  = (const float*)d_in[15];
    const float* convb  = (const float*)d_in[16];
    const float* ln1g   = (const float*)d_in[17];
    const float* ln1b   = (const float*)d_in[18];
    const float* ln2g   = (const float*)d_in[19];
    const float* ln2b   = (const float*)d_in[20];
    float* out = (float*)d_out;

    const size_t NT = (size_t)NTOK * CC;        // 786432
    const size_t WSZ = (size_t)CC * CC;         // 65536
    const size_t CWN = 25ull * CC * CC;         // 1638400

    char* ws = (char*)d_ws;
    size_t off = 0;
    auto alloc = [&](size_t bytes) -> void* {
        void* p = ws + off;
        off = (off + bytes + 255) & ~(size_t)255;
        return p;
    };

    unsigned short* x_bf  = (unsigned short*)alloc(NT * 2);
    unsigned short* wp_q  = (unsigned short*)alloc(WSZ * 2);
    unsigned short* wp_k  = (unsigned short*)alloc(WSZ * 2);
    unsigned short* wp_v  = (unsigned short*)alloc(WSZ * 2);
    unsigned short* wp_qa = (unsigned short*)alloc(WSZ * 2);
    unsigned short* wp_ka = (unsigned short*)alloc(WSZ * 2);
    unsigned short* wp_va = (unsigned short*)alloc(WSZ * 2);
    unsigned short* wp_oa = (unsigned short*)alloc(WSZ * 2);
    unsigned short* cw_p  = (unsigned short*)alloc(CWN * 2);
    unsigned short* q_bf  = (unsigned short*)alloc(NT * 2);
    unsigned short* k_bf  = (unsigned short*)alloc(NT * 2);
    unsigned short* v_bf  = (unsigned short*)alloc(NT * 2);
    unsigned short* kv_bf = (unsigned short*)alloc(NT * 2);
    unsigned short* qa_bf = (unsigned short*)alloc(NT * 2);
    unsigned short* ka_bf = (unsigned short*)alloc(NT * 2);
    unsigned short* va_bf = (unsigned short*)alloc(NT * 2);
    unsigned short* vat_bf = (unsigned short*)alloc(NT * 2);
    unsigned short* ctx_bf = (unsigned short*)alloc(NT * 2);
    float* qloc = (float*)alloc((size_t)BB * PP * CC * 4);
    float* kloc = (float*)alloc((size_t)BB * PP * CC * 4);
    int*   Rbuf = (int*)alloc((size_t)BB * PP * KV * 4);
    float* rout = (float*)alloc(NT * 4);
    float* xres = (float*)alloc(NT * 4);
    float* gbuf = (float*)alloc(NT * 4);

    // conversions / weight packing
    cvt_bf16_kernel<<<(NT + 255) / 256, 256, 0, stream>>>(x, x_bf, (int)NT);
    pack_wB_kernel<<<256, 256, 0, stream>>>(wq,  wp_q);
    pack_wB_kernel<<<256, 256, 0, stream>>>(wk,  wp_k);
    pack_wB_kernel<<<256, 256, 0, stream>>>(wv,  wp_v);
    pack_wB_kernel<<<256, 256, 0, stream>>>(wqa, wp_qa);
    pack_wB_kernel<<<256, 256, 0, stream>>>(wka, wp_ka);
    pack_wB_kernel<<<256, 256, 0, stream>>>(wva, wp_va);
    pack_wB_kernel<<<256, 256, 0, stream>>>(woa, wp_oa);
    pack_cwB_kernel<<<6400, 256, 0, stream>>>(convw, cw_p);

    const dim3 ggrid(NTOK / 64, CC / 16);
    // input projections
    gemm256_kernel<<<ggrid, 128, 0, stream>>>(x_bf, wp_q, bq, q_bf, nullptr);
    gemm256_kernel<<<ggrid, 128, 0, stream>>>(x_bf, wp_k, bk, k_bf, nullptr);
    gemm256_kernel<<<ggrid, 128, 0, stream>>>(x_bf, wp_v, bv, v_bf, nullptr);
    // conv input k_p + v_p
    add_bf16_kernel<<<(NT + 255) / 256, 256, 0, stream>>>(k_bf, v_bf, kv_bf, (int)NT);
    // routing
    block_means_kernel<<<BB * PP, CC, 0, stream>>>(q_bf, k_bf, qloc, kloc);
    routing_topk_kernel<<<1, 32, 0, stream>>>(qloc, kloc, Rbuf);
    // attention projections (on un-gathered tokens; gather happens in attn)
    gemm256_kernel<<<ggrid, 128, 0, stream>>>(q_bf, wp_qa, bqa, qa_bf, nullptr);
    gemm256_kernel<<<ggrid, 128, 0, stream>>>(k_bf, wp_ka, bka, ka_bf, nullptr);
    gemm256_kernel<<<ggrid, 128, 0, stream>>>(v_bf, wp_va, bva, va_bf, nullptr);
    // d-major V for the attention PV stage
    transpose_va_kernel<<<NTOK, 256, 0, stream>>>(va_bf, vat_bf);
    // attention
    attn_kernel<<<dim3(3, NHH, BB * PP), 128, 0, stream>>>(qa_bf, ka_bf, vat_bf,
                                                           Rbuf, ctx_bf);
    // output projection -> routing_out (f32)
    gemm256_kernel<<<ggrid, 128, 0, stream>>>(ctx_bf, wp_oa, boa, nullptr, rout);
    // x_res = LN1(x + routing_out)
    ln_add_kernel<<<NTOK, CC, 0, stream>>>(x, rout, ln1g, ln1b, xres);
    // conv path
    conv_gemm_kernel<<<ggrid, 128, 0, stream>>>(kv_bf, cw_p, convb, gbuf);
    // out = LN2(x_res + g)
    ln_add_kernel<<<NTOK, CC, 0, stream>>>(xres, gbuf, ln2g, ln2b, out);
}
